// FixSetAttention_63513976373590
// MI455X (gfx1250) — matrix operand order repaired and checked by reading, not compiled, not run on hardware
//
#include <hip/hip_runtime.h>
#include <hip/hip_bf16.h>

// ---------------- problem constants ----------------
#define Bq   2
#define Tq   2048
#define Cq   1024
#define NH   16
#define HS   64
#define SSET 1023          // number of dyadic sets
#define SPAD 1024          // padded (set 1023 = zero row, always masked)
#define SCALE 0.125f       // 1/sqrt(64)

typedef __attribute__((ext_vector_type(16))) __bf16 v16bf;
typedef __attribute__((ext_vector_type(8)))  float  v8f;

union BF16x16 { v16bf v; unsigned int u[8]; unsigned short s[16]; };

__device__ __forceinline__ unsigned short f2bf(float f) {
  unsigned int u = __float_as_uint(f);
  u = (u + 0x7FFFu + ((u >> 16) & 1u)) >> 16;   // RNE
  return (unsigned short)u;
}

// A-operand pair index (16-bit A 16x32):
// lanes 0-15: V0-3 K=0..7, V4-7 K=16..23 ; lanes 16-31: +8
__device__ __forceinline__ int pairA(int v, int half) {
  return (v < 4) ? (v + 4 * half) : (v + 4 + 4 * half);
}
// B-operand pair index (16-bit B 32x16): same K order as A:
// lanes 0-15: V0-3 K=0..7, V4-7 K=16..23 ; lanes 16-31: +8
__device__ __forceinline__ int pairB(int v, int half) {
  return (v < 4) ? (v + 4 * half) : (v + 4 + 4 * half);
}

__device__ __forceinline__ v16bf load_a_f32(const float* row, int k0, int half) {
  BF16x16 x;
#pragma unroll
  for (int v = 0; v < 8; ++v) {
    int p = pairA(v, half);
    float2 f = *(const float2*)(row + k0 + 2 * p);
    x.s[2 * v]     = f2bf(f.x);
    x.s[2 * v + 1] = f2bf(f.y);
  }
  return x.v;
}
__device__ __forceinline__ v16bf load_b_f32(const float* row, int k0, int half) {
  BF16x16 x;
#pragma unroll
  for (int v = 0; v < 8; ++v) {
    int p = pairB(v, half);
    float2 f = *(const float2*)(row + k0 + 2 * p);
    x.s[2 * v]     = f2bf(f.x);
    x.s[2 * v + 1] = f2bf(f.y);
  }
  return x.v;
}
__device__ __forceinline__ v16bf load_a_bf16(const unsigned short* row, int k0, int half) {
  BF16x16 x;
  const unsigned int* r32 = (const unsigned int*)(row + k0);
#pragma unroll
  for (int v = 0; v < 8; ++v) x.u[v] = r32[pairA(v, half)];
  return x.v;
}
__device__ __forceinline__ v16bf load_b_bf16(const unsigned short* row, int k0, int half) {
  BF16x16 x;
  const unsigned int* r32 = (const unsigned int*)(row + k0);
#pragma unroll
  for (int v = 0; v < 8; ++v) x.u[v] = r32[pairB(v, half)];
  return x.v;
}

__device__ __forceinline__ v8f wmma_bf16(v16bf a, v16bf b, v8f c) {
  return __builtin_amdgcn_wmma_f32_16x16x32_bf16(false, a, false, b, (short)0, c, false, false);
}

// ---------------- elementwise f32 -> bf16 ----------------
__global__ void cvt_bf16_kernel(const float* __restrict__ in,
                                unsigned short* __restrict__ out, int n) {
  int i = blockIdx.x * blockDim.x + threadIdx.x;
  if (i < n) out[i] = f2bf(in[i]);
}

// ------- register-blocked bf16 WMMA GEMM: C[M,N] = A[M,K] * B[N,K]^T --------
// One wave owns a 32x32 output tile (2x2 subtiles of 16x16): 4 WMMAs per
// (2 A + 2 B) loads -> 16 flop/byte from cache, 4x the WMMA density of a
// single-tile wave.
__global__ void __launch_bounds__(128)
gemm_bf16_nt_kernel(const unsigned short* __restrict__ A,
                    const unsigned short* __restrict__ Bm,
                    float* __restrict__ C, int M, int N, int K) {
  int wave = threadIdx.x >> 5;
  int lane = threadIdx.x & 31;
  int half = lane >> 4, lm = lane & 15;
  int m0 = blockIdx.x * 32;
  int n0 = (blockIdx.y * 4 + wave) * 32;
  const unsigned short* Arow0 = A  + (size_t)(m0 + lm) * K;
  const unsigned short* Arow1 = A  + (size_t)(m0 + 16 + lm) * K;
  const unsigned short* Brow0 = Bm + (size_t)(n0 + lm) * K;
  const unsigned short* Brow1 = Bm + (size_t)(n0 + 16 + lm) * K;
  v8f acc00 = {}, acc01 = {}, acc10 = {}, acc11 = {};
  for (int k0 = 0; k0 < K; k0 += 32) {
    if (k0 + 64 < K) {
      __builtin_prefetch(Arow0 + k0 + 64, 0, 1);
      __builtin_prefetch(Arow1 + k0 + 64, 0, 1);
      __builtin_prefetch(Brow0 + k0 + 64, 0, 1);
      __builtin_prefetch(Brow1 + k0 + 64, 0, 1);
    }
    v16bf a0 = load_a_bf16(Arow0, k0, half);
    v16bf a1 = load_a_bf16(Arow1, k0, half);
    v16bf b0 = load_b_bf16(Brow0, k0, half);
    v16bf b1 = load_b_bf16(Brow1, k0, half);
    acc00 = wmma_bf16(a0, b0, acc00);
    acc01 = wmma_bf16(a0, b1, acc01);
    acc10 = wmma_bf16(a1, b0, acc10);
    acc11 = wmma_bf16(a1, b1, acc11);
  }
#pragma unroll
  for (int r = 0; r < 8; ++r) {
    int m  = m0 + r + 8 * half;
    int m2 = m + 16;
    C[(size_t)m  * N + n0 + lm]      = acc00[r];
    C[(size_t)m  * N + n0 + 16 + lm] = acc01[r];
    C[(size_t)m2 * N + n0 + lm]      = acc10[r];
    C[(size_t)m2 * N + n0 + 16 + lm] = acc11[r];
  }
}

// ---------------- set metadata: r_idx per set + per-32-set-chunk min --------
__global__ void init_sets_kernel(int* __restrict__ r_idx, int* __restrict__ chunk_min) {
  __shared__ int sr[SPAD];
  int s = threadIdx.x;
  int r;
  if (s >= SSET) {
    r = 0x7FFFFFFF;
  } else {
    int off = 0, L = 4;
    while (s >= off + Tq / L) { off += Tq / L; L <<= 1; }
    int i = s - off;
    r = i * L + L - 1;
  }
  r_idx[s] = r;
  sr[s] = r;
  __syncthreads();
  if (s < SPAD / 32) {
    int mn = sr[s * 32];
    for (int j = 1; j < 32; ++j) mn = min(mn, sr[s * 32 + j]);
    chunk_min[s] = mn;
  }
}

// ---------------- level-0 segment sums (L=4) --------------------------------
__global__ void segsum0_kernel(const float* __restrict__ qkv,
                               float* __restrict__ Kseg, float* __restrict__ Vseg) {
  int d = threadIdx.x;                 // 0..63
  int s = blockIdx.x & 511;
  int h = (blockIdx.x >> 9) & 15;
  int b = blockIdx.x >> 13;
  const float* base = qkv + (size_t)(b * Tq + s * 4) * (3 * Cq) + h * HS + d;
  float ks = 0.f, vs = 0.f;
#pragma unroll
  for (int j = 0; j < 4; ++j) {
    ks += base[(size_t)j * (3 * Cq) + Cq];
    vs += base[(size_t)j * (3 * Cq) + 2 * Cq];
  }
  size_t o = ((size_t)((b * NH + h) * SPAD + s)) * HS + d;
  Kseg[o] = ks;
  Vseg[o] = vs;
}

__global__ void segpad_kernel(float* __restrict__ Kseg, float* __restrict__ Vseg) {
  int d = threadIdx.x;
  int bh = blockIdx.x;                 // 0..31
  size_t o = ((size_t)(bh * SPAD + SSET)) * HS + d;
  Kseg[o] = 0.f;
  Vseg[o] = 0.f;
}

// level j from level j-1 (pairwise sums)
__global__ void segup_kernel(float* __restrict__ Kseg, float* __restrict__ Vseg,
                             int off_prev, int off_cur, int count) {
  int d = threadIdx.x;
  int i = blockIdx.x % count;
  int bh = blockIdx.x / count;
  size_t bb  = (size_t)bh * SPAD;
  size_t src = (bb + off_prev + 2 * i) * HS + d;
  size_t dst = (bb + off_cur + i) * HS + d;
  Kseg[dst] = Kseg[src] + Kseg[src + HS];
  Vseg[dst] = Vseg[src] + Vseg[src + HS];
}

// ---------------- tail (length t%4+1) logit + V_tail ------------------------
__global__ void tail_kernel(const float* __restrict__ qkv,
                            float* __restrict__ att_tail, float* __restrict__ V_tail) {
  int idx = blockIdx.x * blockDim.x + threadIdx.x;  // b*nh*T
  int t = idx & (Tq - 1);
  int h = (idx >> 11) & 15;
  int b = idx >> 15;
  int tb = t & ~3;
  const float* qrow = qkv + (size_t)(b * Tq + t) * (3 * Cq) + h * HS;
  float acc = 0.f;
  float* vout = V_tail + (size_t)idx * HS;
  for (int d = 0; d < HS; ++d) {
    float ks = 0.f, vs = 0.f;
    for (int tp = tb; tp <= t; ++tp) {
      const float* r = qkv + (size_t)(b * Tq + tp) * (3 * Cq) + h * HS + d;
      ks += r[Cq];
      vs += r[2 * Cq];
    }
    acc += qrow[d] * ks;
    vout[d] = vs;
  }
  att_tail[idx] = acc * SCALE;
}

// ---------------- flash-style segment attention (1 wave / 16 rows) ----------
__global__ void __launch_bounds__(32)
attn_kernel(const float* __restrict__ qkv,
            const float* __restrict__ Kseg, const float* __restrict__ Vseg,
            const float* __restrict__ att_tail, const float* __restrict__ V_tail,
            const int* __restrict__ r_idx, const int* __restrict__ chunk_min,
            unsigned short* __restrict__ att_out_bf) {
  __shared__ unsigned short Plds[16 * 32];   // probs [m][k]
  __shared__ unsigned short Vlds[32 * 64];   // V chunk [s][d] bf16
  int l = threadIdx.x, half = l >> 4, lm = l & 15;
  int tt = blockIdx.x & 127;
  int h  = (blockIdx.x >> 7) & 15;
  int b  = blockIdx.x >> 11;
  int t0 = tt * 16;
  int bh = b * NH + h;

  // Q tile as A operand (rows m = lm -> t = t0+lm), both K chunks of 32
  const float* qrow = qkv + (size_t)(b * Tq + t0 + lm) * (3 * Cq) + h * HS;
  v16bf qa0 = load_a_f32(qrow, 0, half);
  v16bf qa1 = load_a_f32(qrow, 32, half);

  float rmax[8], rsum[8];
  v8f oacc[4];
#pragma unroll
  for (int r = 0; r < 8; ++r) { rmax[r] = -3.0e38f; rsum[r] = 0.f; }
#pragma unroll
  for (int nt = 0; nt < 4; ++nt) oacc[nt] = (v8f){};

  int tmax = t0 + 15;
  const float* Kbh = Kseg + (size_t)bh * SPAD * HS;
  const float* Vbh = Vseg + (size_t)bh * SPAD * HS;

  for (int c = 0; c < SPAD / 32; ++c) {
    if (chunk_min[c] > tmax) continue;     // uniform skip: fully masked chunk
    int s0 = c * 32;

    // --- logits: two 16-set tiles, K(dims)=64 -> 2 WMMAs each ---
    const float* krow0 = Kbh + (size_t)(s0 + lm) * HS;
    const float* krow1 = Kbh + (size_t)(s0 + 16 + lm) * HS;
    v8f La = {}, Lb = {};
    La = wmma_bf16(qa0, load_b_f32(krow0, 0, half), La);
    La = wmma_bf16(qa1, load_b_f32(krow0, 32, half), La);
    Lb = wmma_bf16(qa0, load_b_f32(krow1, 0, half), Lb);
    Lb = wmma_bf16(qa1, load_b_f32(krow1, 32, half), Lb);

    int ra = r_idx[s0 + lm];
    int rb = r_idx[s0 + 16 + lm];

    // --- online softmax per row (row m lives in one 16-lane half) ---
    float pa[8], pb[8];
#pragma unroll
    for (int r = 0; r < 8; ++r) {
      int trow = t0 + r + 8 * half;
      bool ma = trow >= ra, mb = trow >= rb;
      float lav = ma ? La[r] * SCALE : -3.0e38f;
      float lbv = mb ? Lb[r] * SCALE : -3.0e38f;
      float v = fmaxf(lav, lbv);
#pragma unroll
      for (int o2 = 1; o2 < 16; o2 <<= 1) v = fmaxf(v, __shfl_xor(v, o2, 32));
      float nm = fmaxf(rmax[r], v);
      float f  = __expf(rmax[r] - nm);
      pa[r] = ma ? __expf(lav - nm) : 0.f;
      pb[r] = mb ? __expf(lbv - nm) : 0.f;
      float ps = pa[r] + pb[r];
#pragma unroll
      for (int o2 = 1; o2 < 16; o2 <<= 1) ps += __shfl_xor(ps, o2, 32);
      rsum[r] = rsum[r] * f + ps;
      rmax[r] = nm;
      oacc[0][r] *= f; oacc[1][r] *= f; oacc[2][r] *= f; oacc[3][r] *= f;
    }

    // --- probs -> LDS (row-major [m][32]) for A-operand reload ---
#pragma unroll
    for (int r = 0; r < 8; ++r) {
      int m = r + 8 * half;
      Plds[m * 32 + lm]      = f2bf(pa[r]);
      Plds[m * 32 + 16 + lm] = f2bf(pb[r]);
    }
    // --- stage V chunk [32 s][64 d] into LDS as bf16 (1 row / lane) ---
    {
      const float* vrow = Vbh + (size_t)(s0 + l) * HS;
      unsigned int* vl = (unsigned int*)&Vlds[l * 64];
#pragma unroll
      for (int j = 0; j < 32; ++j) {
        float2 f = *(const float2*)(vrow + 2 * j);
        vl[j] = (unsigned int)f2bf(f.x) | ((unsigned int)f2bf(f.y) << 16);
      }
    }
    __syncthreads();

    // --- P (16x32) as A operand ---
    v16bf pA;
    {
      BF16x16 x;
      const unsigned int* pr = (const unsigned int*)&Plds[lm * 32];
#pragma unroll
      for (int v = 0; v < 8; ++v) x.u[v] = pr[pairA(v, half)];
      pA = x.v;
    }
    // --- out += P @ V : 4 dim-tiles of 16, B operand = V^T from LDS ---
#pragma unroll
    for (int nt = 0; nt < 4; ++nt) {
      BF16x16 xb;
      int d = nt * 16 + lm;
#pragma unroll
      for (int v = 0; v < 8; ++v) {
        int k0 = 2 * pairB(v, half);
        xb.u[v] = (unsigned int)Vlds[k0 * 64 + d] |
                  ((unsigned int)Vlds[(k0 + 1) * 64 + d] << 16);
      }
      oacc[nt] = wmma_bf16(pA, xb.v, oacc[nt]);
    }
    __syncthreads();
  }

  // --- fold in the tail column, normalize, emit bf16 for final GEMM ---
  const float* atl = att_tail + (size_t)bh * Tq;
  const float* vtl = V_tail + (size_t)bh * Tq * HS;
#pragma unroll
  for (int r = 0; r < 8; ++r) {
    int t = t0 + r + 8 * half;
    float lt = atl[t];
    float nm = fmaxf(rmax[r], lt);
    float f  = __expf(rmax[r] - nm);
    float pt = __expf(lt - nm);
    rsum[r] = rsum[r] * f + pt;
#pragma unroll
    for (int nt = 0; nt < 4; ++nt) {
      float vt = vtl[(size_t)t * HS + nt * 16 + lm];
      oacc[nt][r] = oacc[nt][r] * f + pt * vt;
    }
    float inv = 1.f / rsum[r];
#pragma unroll
    for (int nt = 0; nt < 4; ++nt) {
      att_out_bf[(size_t)(b * Tq + t) * Cq + h * HS + nt * 16 + lm] =
          f2bf(oacc[nt][r] * inv);
    }
  }
}

// ---------------- host-side orchestration ------------------------------------
extern "C" void kernel_launch(void* const* d_in, const int* in_sizes, int n_in,
                              void* d_out, int out_size, void* d_ws, size_t ws_size,
                              hipStream_t stream) {
  (void)in_sizes; (void)n_in; (void)out_size; (void)ws_size;
  const float* x      = (const float*)d_in[0];   // [B,T,C]
  const float* W_attn = (const float*)d_in[1];   // [3C,C]
  const float* W_proj = (const float*)d_in[2];   // [C,C]
  float* out = (float*)d_out;                    // [B,T,C]

  char* ws = (char*)d_ws;
  size_t off = 0;
  auto take = [&](size_t bytes) -> char* {
    char* p = ws + off;
    off += (bytes + 255) & ~(size_t)255;
    return p;
  };
  float*          qkv        = (float*)take((size_t)Bq * Tq * 3 * Cq * 4);
  float*          Kseg       = (float*)take((size_t)Bq * NH * SPAD * HS * 4);
  float*          Vseg       = (float*)take((size_t)Bq * NH * SPAD * HS * 4);
  float*          att_tail   = (float*)take((size_t)Bq * NH * Tq * 4);
  float*          V_tail     = (float*)take((size_t)Bq * NH * Tq * HS * 4);
  unsigned short* xb         = (unsigned short*)take((size_t)Bq * Tq * Cq * 2);
  unsigned short* Wab        = (unsigned short*)take((size_t)3 * Cq * Cq * 2);
  unsigned short* Wpb        = (unsigned short*)take((size_t)Cq * Cq * 2);
  unsigned short* att_out_bf = (unsigned short*)take((size_t)Bq * Tq * Cq * 2);
  int*            r_idx      = (int*)take(SPAD * 4);
  int*            chunk_min  = (int*)take((SPAD / 32) * 4);

  const int nx = Bq * Tq * Cq;       // 4194304
  const int na = 3 * Cq * Cq;        // 3145728
  const int np = Cq * Cq;            // 1048576
  cvt_bf16_kernel<<<(nx + 255) / 256, 256, 0, stream>>>(x, xb, nx);
  cvt_bf16_kernel<<<(na + 255) / 256, 256, 0, stream>>>(W_attn, Wab, na);
  cvt_bf16_kernel<<<(np + 255) / 256, 256, 0, stream>>>(W_proj, Wpb, np);

  // qkv = x @ W_attn^T   (M=4096, N=3072, K=1024); 32x32 tile per wave,
  // 4 waves per block -> block covers 32(M) x 128(N)
  gemm_bf16_nt_kernel<<<dim3(Bq * Tq / 32, 3 * Cq / 128), 128, 0, stream>>>(
      xb, Wab, qkv, Bq * Tq, 3 * Cq, Cq);

  init_sets_kernel<<<1, SPAD, 0, stream>>>(r_idx, chunk_min);
  segsum0_kernel<<<Bq * NH * 512, HS, 0, stream>>>(qkv, Kseg, Vseg);
  segpad_kernel<<<Bq * NH, HS, 0, stream>>>(Kseg, Vseg);
  for (int j = 1; j <= 9; ++j) {
    int off_prev = SPAD - (2048 >> j);
    int off_cur  = SPAD - (1024 >> j);
    int count    = 512 >> j;
    segup_kernel<<<Bq * NH * count, HS, 0, stream>>>(Kseg, Vseg, off_prev, off_cur, count);
  }

  tail_kernel<<<(Bq * NH * Tq) / 256, 256, 0, stream>>>(qkv, att_tail, V_tail);

  attn_kernel<<<Bq * NH * (Tq / 16), 32, 0, stream>>>(
      qkv, Kseg, Vseg, att_tail, V_tail, r_idx, chunk_min, att_out_bf);

  // out = att_out @ W_proj^T  (M=4096, N=1024, K=1024)
  gemm_bf16_nt_kernel<<<dim3(Bq * Tq / 32, Cq / 128), 128, 0, stream>>>(
      att_out_bf, Wpb, out, Bq * Tq, Cq, Cq);
}
